// TextSelfAttentionModule_50929722196591
// MI455X (gfx1250) — matrix operand order repaired and checked by reading, not compiled, not run on hardware
//
#include <hip/hip_runtime.h>
#include <hip/hip_bf16.h>

// ---------------------------------------------------------------------------
// Problem constants (from reference): B=64, S=512, E=256, H=512, V=32000, K=8
// ---------------------------------------------------------------------------
#define SLEN   512
#define BATCH  64
#define EDIM   256
#define HDIM   512
#define KHEADS 8
#define G4H    2048   // 4*H

typedef __attribute__((ext_vector_type(16))) _Float16 v16h;
typedef __attribute__((ext_vector_type(8)))  _Float16 v8h;
typedef __attribute__((ext_vector_type(4)))  _Float16 v4h;
typedef __attribute__((ext_vector_type(8)))  float    v8f;
typedef __attribute__((ext_vector_type(4)))  unsigned int v4u;
typedef __attribute__((ext_vector_type(8)))  int      v8i;
typedef __attribute__((ext_vector_type(4)))  int      v4i;

// ---------------------------------------------------------------------------
// WMMA helpers (CDNA5: v_wmma_f32_16x16x32_f16, wave32)
// ---------------------------------------------------------------------------
__device__ __forceinline__ v8f wmma_f16f32(v16h a, v16h b, v8f c) {
  return __builtin_amdgcn_wmma_f32_16x16x32_f16(false, a, false, b, (short)0, c,
                                                false, false);
}

// A fragment from row-major global memory (lda elems). ISA 7.12.2 16-bit A:
// lanes 0-15: M=lane,  elems[0..7]=K k0..+7,  elems[8..15]=K k0+16..+23
// lanes 16-31: M=lane-16, K ranges shifted by +8.
__device__ __forceinline__ v16h load_a_frag(const _Float16* __restrict__ A, int lda,
                                            int row0, int k0, int lane) {
  int m    = lane & 15;
  int half = lane >> 4;
  const _Float16* p = A + (size_t)(row0 + m) * lda + k0 + half * 8;
  union { v16h v; v8h h[2]; } u;
  u.h[0] = *(const v8h*)(p);
  u.h[1] = *(const v8h*)(p + 16);
  return u.v;
}

// B fragment: B[k][n] = W[n][k], W row-major [N x ldw].
// lanes 0-15: N=lane, K k0..+7, k0+16..+23 ; lanes 16-31: N=lane-16, K ranges shifted by +8
__device__ __forceinline__ v16h load_b_frag(const _Float16* __restrict__ W, int ldw,
                                            int n0, int k0, int lane) {
  int n    = lane & 15;
  int half = lane >> 4;
  const _Float16* p = W + (size_t)(n0 + n) * ldw + k0 + half * 8;
  union { v16h v; v8h h[2]; } u;
  u.h[0] = *(const v8h*)(p);
  u.h[1] = *(const v8h*)(p + 16);
  return u.v;
}

// ---- LDS-resident fragments, row pitch 40 halves (80 B: conflict-free) ----
#define LDSPITCH 40

__device__ __forceinline__ v16h lds_a_frag(const _Float16* base, int row0, int lane) {
  int m = lane & 15, half = lane >> 4;
  const _Float16* p = base + (row0 + m) * LDSPITCH + half * 8;
  union { v16h v; v8h h[2]; } u;
  u.h[0] = *(const v8h*)(p);
  u.h[1] = *(const v8h*)(p + 16);
  return u.v;
}

__device__ __forceinline__ v16h lds_b_frag(const _Float16* base, int row0, int lane) {
  int n = lane & 15, half = lane >> 4;
  const _Float16* p = base + (row0 + n) * LDSPITCH + half * 8;
  union { v16h v; v8h h[2]; } u;
  u.h[0] = *(const v8h*)(p);
  u.h[1] = *(const v8h*)(p + 16);
  return u.v;
}

// ---------------------------------------------------------------------------
// Tensor Data Mover: 2D tile (32 x 128 rows, f16) global -> LDS with padding
// (pad 4 DWORDs after every 16 DWORDs => LDS row pitch 80 B). D# per ISA ch.8.
// 6-arg builtin: (uint32x4 g0, int32x8 g1, int32x4, int32x4, int32x8, i32 cpol)
// ---------------------------------------------------------------------------
__device__ __forceinline__ void tdm_load_2d(const _Float16* gptr, unsigned lds_off,
                                            int tensor_d0, int tensor_d1,
                                            int row_stride) {
  unsigned long long ga = (unsigned long long)(size_t)gptr;
  v4u g0;
  g0.x = 1u;                                            // count=1, user desc
  g0.y = lds_off;                                       // lds_addr (bytes)
  g0.z = (unsigned)ga;                                  // global_addr[31:0]
  g0.w = (unsigned)((ga >> 32) & 0x01FFFFFFu) | 0x80000000u;  // addr[56:32]|type=2
  v8i g1;
  g1[0] = (1 << 16)      // data_size = 2 bytes
        | (1 << 20)      // pad_enable
        | (3 << 22)      // pad_interval: 16 DWORDs (one 64B row)
        | (3 << 25);     // pad_amount: 4 DWORDs (16 B)
  g1[1] = (tensor_d0 & 0xFFFF) << 16;                               // dim0[15:0]
  g1[2] = ((tensor_d0 >> 16) & 0xFFFF) | ((tensor_d1 & 0xFFFF) << 16);
  g1[3] = ((tensor_d1 >> 16) & 0xFFFF) | (32 << 16);                // tile_dim0=32
  g1[4] = 128;                                                      // tile_dim1=128
  g1[5] = row_stride;                                               // dim0_stride lo
  g1[6] = 0;
  g1[7] = 0;
  v4i z4 = {0, 0, 0, 0};
  v8i z8 = {0, 0, 0, 0, 0, 0, 0, 0};
  __builtin_amdgcn_tensor_load_to_lds(g0, g1, z4, z4, z8, 0);
}

__device__ __forceinline__ float sigmf(float x) { return 1.0f / (1.0f + __expf(-x)); }

// ---------------------------------------------------------------------------
// Device-wide arrive/generation barrier (persistent LSTM step sync)
// ---------------------------------------------------------------------------
__device__ __forceinline__ void grid_barrier(unsigned* cnt, unsigned* gen,
                                             unsigned nblocks) {
  __syncthreads();
  if (threadIdx.x == 0) {
    __threadfence();
    unsigned g = __hip_atomic_load(gen, __ATOMIC_ACQUIRE, __HIP_MEMORY_SCOPE_AGENT);
    unsigned a = __hip_atomic_fetch_add(cnt, 1u, __ATOMIC_ACQ_REL,
                                        __HIP_MEMORY_SCOPE_AGENT);
    if (a == nblocks - 1) {
      __hip_atomic_store(cnt, 0u, __ATOMIC_RELAXED, __HIP_MEMORY_SCOPE_AGENT);
      __hip_atomic_fetch_add(gen, 1u, __ATOMIC_ACQ_REL, __HIP_MEMORY_SCOPE_AGENT);
    } else {
      while (__hip_atomic_load(gen, __ATOMIC_ACQUIRE, __HIP_MEMORY_SCOPE_AGENT) == g)
        __builtin_amdgcn_s_sleep(2);
    }
    __threadfence();
  }
  __syncthreads();
}

// ---------------------------------------------------------------------------
// 1) f32 -> f16 weight conversion
// ---------------------------------------------------------------------------
__global__ __launch_bounds__(256) void cvt_f16_kernel(const float* __restrict__ src,
                                                      _Float16* __restrict__ dst, int n) {
  int i = blockIdx.x * 256 + threadIdx.x;
  if (i < n) dst[i] = (_Float16)src[i];
}

// ---------------------------------------------------------------------------
// 2) Embedding gather + per-sample reversal, packed to f16 as [S, B, E]
// ---------------------------------------------------------------------------
__global__ __launch_bounds__(128) void embed_pack_kernel(
    const int* __restrict__ tokens, const int* __restrict__ lengths,
    const float* __restrict__ emb,
    _Float16* __restrict__ xf, _Float16* __restrict__ xb) {
  int bs = blockIdx.x;
  int b  = bs >> 9;
  int s  = bs & (SLEN - 1);
  int len = lengths[b];
  int tf = (s < len) ? tokens[b * SLEN + s] : 0;
  int tb = (s < len) ? tokens[b * SLEN + (len - 1 - s)] : 0;
  const float* ef = emb + (size_t)tf * EDIM;
  const float* eb = emb + (size_t)tb * EDIM;
  _Float16* of = xf + ((size_t)s * BATCH + b) * EDIM;
  _Float16* ob = xb + ((size_t)s * BATCH + b) * EDIM;
  for (int e = threadIdx.x; e < EDIM; e += 128) {
    of[e] = (_Float16)ef[e];
    ob[e] = (_Float16)eb[e];
  }
}

// ---------------------------------------------------------------------------
// 3) Persistent sequential LSTM (one direction per launch).
//    gates[64x2048] = [x_s | h] @ [Wih | Whh]^T via WMMA; B fragments are
//    re-read from global on each of the 512 steps. A-fragments are fetched in groups of 4
//    distinct registers so the loads of a group issue back-to-back and the
//    recurrence pays ~6 memory latencies per step instead of 24.
// ---------------------------------------------------------------------------
__global__ __launch_bounds__(128) void lstm_seq_kernel(
    const _Float16* __restrict__ x16,    // [S, B, E]
    const _Float16* __restrict__ Wih16,  // [4H, E]
    const _Float16* __restrict__ Whh16,  // [4H, H]
    const float* __restrict__ bih, const float* __restrict__ bhh,
    const int* __restrict__ lengths,
    float* __restrict__ gates,           // [B, 4H]
    float* __restrict__ cst,             // [B, H]
    _Float16* __restrict__ h16,          // [B, H]
    _Float16* __restrict__ joined,       // [B, S, 2H]
    unsigned* __restrict__ bar,
    int dir) {
  const int lane   = threadIdx.x & 31;
  const int wave   = threadIdx.x >> 5;
  const int n_tile = blockIdx.x;
  const int tid    = blockIdx.x * 128 + threadIdx.x;

  for (int s = 0; s < SLEN; ++s) {
    v8f acc = {};
    const _Float16* xs = x16 + (size_t)s * (BATCH * EDIM);

    // 24 chained wmmas over K = [E=256 | H=512]; 6 groups of 4 batched loads.
#pragma unroll
    for (int g = 0; g < 6; ++g) {
      v16h a4[4];
#pragma unroll
      for (int u = 0; u < 4; ++u) {
        int t = g * 4 + u;
        a4[u] = (t < 8)
                    ? load_a_frag(xs, EDIM, wave * 16, t * 32, lane)
                    : load_a_frag(h16, HDIM, wave * 16, (t - 8) * 32, lane);
      }
#pragma unroll
      for (int u = 0; u < 4; ++u) {
        int t = g * 4 + u;
        v16h bf = (t < 8)
                      ? load_b_frag(Wih16, EDIM, n_tile * 16, t * 32, lane)
                      : load_b_frag(Whh16, HDIM, n_tile * 16, (t - 8) * 32, lane);
        acc = wmma_f16f32(a4[u], bf, acc);
      }
    }
    {
      int n  = n_tile * 16 + (lane & 15);
      int mb = wave * 16 + ((lane >> 4) * 8);
#pragma unroll
      for (int r = 0; r < 8; ++r)
        gates[(size_t)(mb + r) * G4H + n] = acc[r];
    }
    grid_barrier(bar, bar + 1, gridDim.x);

    // activation phase (torch gate order i,f,g,o)
    for (int idx = tid; idx < BATCH * HDIM; idx += 128 * 128) {
      int b = idx >> 9;
      int h = idx & (HDIM - 1);
      const float* g = gates + (size_t)b * G4H;
      float gi = g[h]            + bih[h]            + bhh[h];
      float gf = g[HDIM + h]     + bih[HDIM + h]     + bhh[HDIM + h];
      float gg = g[2 * HDIM + h] + bih[2 * HDIM + h] + bhh[2 * HDIM + h];
      float go = g[3 * HDIM + h] + bih[3 * HDIM + h] + bhh[3 * HDIM + h];
      float c  = cst[idx];
      c = sigmf(gf) * c + sigmf(gi) * tanhf(gg);
      float hv = sigmf(go) * tanhf(c);
      cst[idx] = c;
      h16[idx] = (_Float16)hv;
      int len = lengths[b];
      if (s < len) {
        size_t pos = (dir == 0) ? (size_t)s : (size_t)(len - 1 - s);
        joined[((size_t)b * SLEN + pos) * (2 * HDIM) + (dir ? HDIM : 0) + h] =
            (_Float16)hv;
      }
    }
    // warm the WGP cache with next step's 32 KB x-slab (global_prefetch_b8)
    if (s + 1 < SLEN && tid < 256) {
      const _Float16* xn = x16 + (size_t)(s + 1) * (BATCH * EDIM);
      __builtin_prefetch(xn + (size_t)tid * 64, 0, 3);
    }
    grid_barrier(bar, bar + 1, gridDim.x);
  }
}

// ---------------------------------------------------------------------------
// 4) scores init: scores[b,s,k] = b2[k]
// ---------------------------------------------------------------------------
__global__ __launch_bounds__(256) void init_scores_kernel(const float* __restrict__ b2,
                                                          float* __restrict__ scores,
                                                          int n) {
  int i = blockIdx.x * 256 + threadIdx.x;
  if (i < n) scores[i] = b2[i & (KHEADS - 1)];
}

// ---------------------------------------------------------------------------
// 5) Attention scores GEMM, TDM double-buffered + register-blocked.
//    Grid: (mblock=256, nblock=4, head=8); block = 256 threads = 8 waves.
//    Macro tile 128(M) x 128(N); wave tile 64x32 = 4x2 accumulators.
//    Per k-step: TDM fetches A[128x32] and B[128x32] into padded LDS.
//    Fused epilogue: tanh(+b1) * W2, in-wave N-reduce, atomicAdd -> scores.
// ---------------------------------------------------------------------------
__global__ __launch_bounds__(256) void attn_scores_kernel(
    const _Float16* __restrict__ joined,  // [32768, 1024]
    const _Float16* __restrict__ W1_16,   // [8, 512, 1024]
    const float* __restrict__ b1,         // [8, 512]
    const float* __restrict__ W2,         // [8, 512]
    float* __restrict__ scores) {         // [32768, 8]
  __shared__ _Float16 sh[4][128 * LDSPITCH];   // A0,A1,B0,B1 : 40 KB
  const unsigned bufBytes = 128 * LDSPITCH * 2;

  const int lane = threadIdx.x & 31;
  const int wave = threadIdx.x >> 5;
  const int mblock = blockIdx.x;
  const int nblock = blockIdx.y;
  const int head   = blockIdx.z;

  const _Float16* Abase = joined + (size_t)mblock * 128 * (2 * HDIM);
  const _Float16* Bbase = W1_16 + (size_t)head * HDIM * (2 * HDIM) +
                          (size_t)nblock * 128 * (2 * HDIM);
  const int wm = (wave >> 2) * 64;   // 0 / 64
  const int wn = (wave & 3) * 32;    // 0 / 32 / 64 / 96

  v8f acc[4][2] = {};

  if (wave == 0) {
    tdm_load_2d(Abase, 0u, 2 * HDIM, BATCH * SLEN, 2 * HDIM);
    tdm_load_2d(Bbase, 2u * bufBytes, 2 * HDIM, HDIM, 2 * HDIM);
  }
  for (int k = 0; k < 32; ++k) {
    const int cur = k & 1;
    if (wave == 0) {
      if (k + 1 < 32) {
        tdm_load_2d(Abase + (k + 1) * 32, (unsigned)(1 - cur) * bufBytes,
                    2 * HDIM, BATCH * SLEN, 2 * HDIM);
        tdm_load_2d(Bbase + (k + 1) * 32, (unsigned)(3 - cur) * bufBytes,
                    2 * HDIM, HDIM, 2 * HDIM);
        __builtin_amdgcn_s_wait_tensorcnt((short)2);   // step-k pair done
      } else {
        __builtin_amdgcn_s_wait_tensorcnt((short)0);
      }
    }
    __syncthreads();
    const _Float16* Ash = sh[cur];
    const _Float16* Bsh = sh[2 + cur];
    v16h bfr0 = lds_b_frag(Bsh, wn, lane);
    v16h bfr1 = lds_b_frag(Bsh, wn + 16, lane);
#pragma unroll
    for (int i = 0; i < 4; ++i) {
      v16h af = lds_a_frag(Ash, wm + i * 16, lane);
      acc[i][0] = wmma_f16f32(af, bfr0, acc[i][0]);
      acc[i][1] = wmma_f16f32(af, bfr1, acc[i][1]);
    }
    __syncthreads();   // everyone done with buf[cur] before it is refilled
  }

  // ---- fused epilogue ----
#pragma unroll
  for (int j = 0; j < 2; ++j) {
    int n_glob = nblock * 128 + wn + j * 16 + (lane & 15);
    float b1v = b1[head * HDIM + n_glob];
    float w2v = W2[head * HDIM + n_glob];
#pragma unroll
    for (int i = 0; i < 4; ++i) {
      float v[8];
#pragma unroll
      for (int r = 0; r < 8; ++r) v[r] = tanhf(acc[i][j][r] + b1v) * w2v;
#pragma unroll
      for (int off = 1; off < 16; off <<= 1)
#pragma unroll
        for (int r = 0; r < 8; ++r) v[r] += __shfl_xor(v[r], off, 32);
      if ((lane & 15) == 0) {
        int mb = mblock * 128 + wm + i * 16 + ((lane >> 4) * 8);
#pragma unroll
        for (int r = 0; r < 8; ++r)
          atomicAdd(&scores[(size_t)(mb + r) * KHEADS + head], v[r]);
      }
    }
  }
}

// ---------------------------------------------------------------------------
// 6) Masked softmax over the time axis, per (b, k)
// ---------------------------------------------------------------------------
__global__ __launch_bounds__(256) void softmax_kernel(
    const float* __restrict__ scores, const int* __restrict__ lengths,
    float* __restrict__ probs) {
  int b = blockIdx.x >> 3;
  int k = blockIdx.x & 7;
  int len = lengths[b];
  const float* sc = scores + (size_t)b * SLEN * KHEADS + k;
  float* pr       = probs  + (size_t)b * SLEN * KHEADS + k;
  __shared__ float red[256];
  int t = threadIdx.x;
  float m = -3.0e38f;
  for (int s = t; s < len; s += 256) m = fmaxf(m, sc[(size_t)s * KHEADS]);
  red[t] = m; __syncthreads();
  for (int o = 128; o > 0; o >>= 1) {
    if (t < o) red[t] = fmaxf(red[t], red[t + o]);
    __syncthreads();
  }
  float mx = red[0]; __syncthreads();
  float sum = 0.f;
  for (int s = t; s < SLEN; s += 256) {
    float e = 0.f;
    if (s < len) e = __expf(sc[(size_t)s * KHEADS] - mx);
    pr[(size_t)s * KHEADS] = e;
    sum += e;
  }
  red[t] = sum; __syncthreads();
  for (int o = 128; o > 0; o >>= 1) {
    if (t < o) red[t] += red[t + o];
    __syncthreads();
  }
  float inv = 1.0f / red[0];
  for (int s = t; s < len; s += 256) pr[(size_t)s * KHEADS] *= inv;
}

// ---------------------------------------------------------------------------
// 7) Pooling: out[b, k*2H + d] = sum_s probs[b,s,k] * joined[b,s,d]
// ---------------------------------------------------------------------------
__global__ __launch_bounds__(256) void pool_kernel(
    const float* __restrict__ probs, const _Float16* __restrict__ joined,
    const int* __restrict__ lengths, float* __restrict__ out) {
  int b = blockIdx.x >> 3;
  int k = blockIdx.x & 7;
  int len = lengths[b];
  int d = threadIdx.x * 4;
  float a0 = 0.f, a1 = 0.f, a2 = 0.f, a3 = 0.f;
  for (int s = 0; s < len; ++s) {
    float pv = probs[((size_t)b * SLEN + s) * KHEADS + k];
    v4h jv = *(const v4h*)(joined + ((size_t)b * SLEN + s) * (2 * HDIM) + d);
    a0 += pv * (float)jv.x; a1 += pv * (float)jv.y;
    a2 += pv * (float)jv.z; a3 += pv * (float)jv.w;
  }
  float* o = out + (size_t)b * (KHEADS * 2 * HDIM) + (size_t)k * (2 * HDIM) + d;
  o[0] = a0; o[1] = a1; o[2] = a2; o[3] = a3;
}

// ---------------------------------------------------------------------------
// Host-side orchestration
// ---------------------------------------------------------------------------
extern "C" void kernel_launch(void* const* d_in, const int* in_sizes, int n_in,
                              void* d_out, int out_size, void* d_ws, size_t ws_size,
                              hipStream_t stream) {
  const int*   tokens  = (const int*)d_in[0];
  const int*   lengths = (const int*)d_in[1];
  const float* emb     = (const float*)d_in[2];
  const float* Wih_f   = (const float*)d_in[3];
  const float* Whh_f   = (const float*)d_in[4];
  const float* bih_f   = (const float*)d_in[5];
  const float* bhh_f   = (const float*)d_in[6];
  const float* Wih_b   = (const float*)d_in[7];
  const float* Whh_b   = (const float*)d_in[8];
  const float* bih_b   = (const float*)d_in[9];
  const float* bhh_b   = (const float*)d_in[10];
  const float* W1      = (const float*)d_in[11];
  const float* b1      = (const float*)d_in[12];
  const float* W2      = (const float*)d_in[13];
  const float* b2      = (const float*)d_in[14];
  float* out = (float*)d_out;

  char* p = (char*)d_ws;
  auto alloc = [&](size_t bytes) -> void* {
    void* r = (void*)p;
    p += (bytes + 255) & ~(size_t)255;
    return r;
  };
  _Float16* x16_f   = (_Float16*)alloc((size_t)SLEN * BATCH * EDIM * 2);
  _Float16* x16_b   = (_Float16*)alloc((size_t)SLEN * BATCH * EDIM * 2);
  _Float16* Wih16f  = (_Float16*)alloc((size_t)G4H * EDIM * 2);
  _Float16* Whh16f  = (_Float16*)alloc((size_t)G4H * HDIM * 2);
  _Float16* Wih16b  = (_Float16*)alloc((size_t)G4H * EDIM * 2);
  _Float16* Whh16b  = (_Float16*)alloc((size_t)G4H * HDIM * 2);
  _Float16* W1_16   = (_Float16*)alloc((size_t)KHEADS * HDIM * (2 * HDIM) * 2);
  _Float16* joined  = (_Float16*)alloc((size_t)BATCH * SLEN * (2 * HDIM) * 2);
  float*    gates_f = (float*)alloc((size_t)BATCH * G4H * 4);
  float*    gates_b = (float*)alloc((size_t)BATCH * G4H * 4);
  float*    cst_f   = (float*)alloc((size_t)BATCH * HDIM * 4);
  float*    cst_b   = (float*)alloc((size_t)BATCH * HDIM * 4);
  _Float16* h16_f   = (_Float16*)alloc((size_t)BATCH * HDIM * 2);
  _Float16* h16_b   = (_Float16*)alloc((size_t)BATCH * HDIM * 2);
  float*    scores  = (float*)alloc((size_t)BATCH * SLEN * KHEADS * 4);
  float*    probs   = (float*)alloc((size_t)BATCH * SLEN * KHEADS * 4);
  unsigned* bar_f   = (unsigned*)alloc(256);
  unsigned* bar_b   = (unsigned*)alloc(256);

  auto cvt = [&](const float* s, _Float16* d, int n) {
    cvt_f16_kernel<<<(n + 255) / 256, 256, 0, stream>>>(s, d, n);
  };
  cvt(Wih_f, Wih16f, G4H * EDIM);
  cvt(Whh_f, Whh16f, G4H * HDIM);
  cvt(Wih_b, Wih16b, G4H * EDIM);
  cvt(Whh_b, Whh16b, G4H * HDIM);
  cvt(W1, W1_16, KHEADS * HDIM * 2 * HDIM);

  embed_pack_kernel<<<BATCH * SLEN, 128, 0, stream>>>(tokens, lengths, emb,
                                                      x16_f, x16_b);

  (void)hipMemsetAsync(joined, 0, (size_t)BATCH * SLEN * (2 * HDIM) * 2, stream);
  (void)hipMemsetAsync(cst_f, 0, (size_t)BATCH * HDIM * 4, stream);
  (void)hipMemsetAsync(cst_b, 0, (size_t)BATCH * HDIM * 4, stream);
  (void)hipMemsetAsync(h16_f, 0, (size_t)BATCH * HDIM * 2, stream);
  (void)hipMemsetAsync(h16_b, 0, (size_t)BATCH * HDIM * 2, stream);
  (void)hipMemsetAsync(bar_f, 0, 256, stream);
  (void)hipMemsetAsync(bar_b, 0, 256, stream);

  lstm_seq_kernel<<<128, 128, 0, stream>>>(x16_f, Wih16f, Whh16f, bih_f, bhh_f,
                                           lengths, gates_f, cst_f, h16_f, joined,
                                           bar_f, /*dir=*/0);
  lstm_seq_kernel<<<128, 128, 0, stream>>>(x16_b, Wih16b, Whh16b, bih_b, bhh_b,
                                           lengths, gates_b, cst_b, h16_b, joined,
                                           bar_b, /*dir=*/1);

  int nscore = BATCH * SLEN * KHEADS;
  init_scores_kernel<<<(nscore + 255) / 256, 256, 0, stream>>>(b2, scores, nscore);

  dim3 agrid(BATCH * SLEN / 128, HDIM / 128, KHEADS);   // (256, 4, 8)
  attn_scores_kernel<<<agrid, 256, 0, stream>>>(joined, W1_16, b1, W2, scores);

  softmax_kernel<<<BATCH * KHEADS, 256, 0, stream>>>(scores, lengths, probs);
  pool_kernel<<<BATCH * KHEADS, 256, 0, stream>>>(probs, joined, lengths, out);
}
